// MultiHeadAttention_12481174962184
// MI455X (gfx1250) — matrix operand order repaired and checked by reading, not compiled, not run on hardware
//
#include <hip/hip_runtime.h>
#include <cstdint>
#include <cstddef>

// ---------------------------------------------------------------------------
// CDNA5 (gfx1250) bf16 WMMA fragment types / helpers
// ---------------------------------------------------------------------------
typedef __attribute__((ext_vector_type(16))) __bf16 v16bf;
typedef __attribute__((ext_vector_type(8)))  float  v8f;

union Frag {
  v16bf          bf;
  uint4          q[2];
  uint32_t       w[8];
  unsigned short us[16];
};

__device__ __forceinline__ unsigned short f32_to_bf16u(float f) {
  uint32_t u = __builtin_bit_cast(uint32_t, f);
  uint32_t r = u + 0x7FFFu + ((u >> 16) & 1u);   // round-to-nearest-even
  return (unsigned short)(r >> 16);
}

#if __has_builtin(__builtin_amdgcn_cvt_pk_bf16_f32)
__device__ __forceinline__ uint32_t pack2_bf16(float lo, float hi) {
  typedef __attribute__((ext_vector_type(2))) __bf16 v2bf;
  v2bf r = __builtin_amdgcn_cvt_pk_bf16_f32(lo, hi);
  return __builtin_bit_cast(uint32_t, r);
}
#else
__device__ __forceinline__ uint32_t pack2_bf16(float lo, float hi) {
  return (uint32_t)f32_to_bf16u(lo) | ((uint32_t)f32_to_bf16u(hi) << 16);
}
#endif

// SWAPX16 within wave32: xor_mask=0x10, and_mask=0x1f  -> offset 0x401F
__device__ __forceinline__ float swap16f(float x) {
  int v = __builtin_amdgcn_ds_swizzle(__builtin_bit_cast(int, x), 0x401F);
  return __builtin_bit_cast(float, v);
}
__device__ __forceinline__ v8f wmma_bf16(const Frag& a, const Frag& b, v8f c) {
  return __builtin_amdgcn_wmma_f32_16x16x32_bf16(
      /*neg_a=*/false, a.bf, /*neg_b=*/false, b.bf,
      /*c_mod=*/(short)0, c, /*reuse_a=*/false, /*reuse_b=*/false);
}

// ---------------------------------------------------------------------------
// Tensor Data Mover (TDM): DMA a 2D bf16 tile into LDS with row padding that
// matches the 80B (LSTR=40 ushort) bank-spread LDS layout:
//   row payload = 64B = 16 DWORDs  -> pad_interval = 3
//   pad         = 16B =  4 DWORDs  -> pad_amount   = 3
// This toolchain exposes the 6-arg builtin form:
//   (uint32x4 g0, int32x8 g1, int32x4 g2, int32x4 g3, int32x8 g4, i32 cpol)
// Helpers are always DECLARED (host pass must resolve names inside discarded
// if-constexpr branches); the builtin calls are device-pass-only.
// ---------------------------------------------------------------------------
typedef __attribute__((ext_vector_type(4))) unsigned int u32x4;
typedef __attribute__((ext_vector_type(8))) int          i32x8;
typedef __attribute__((ext_vector_type(4))) int          i32x4;

#if defined(__HIP_DEVICE_COMPILE__)
  #if __has_builtin(__builtin_amdgcn_tensor_load_to_lds) && \
      __has_builtin(__builtin_amdgcn_s_wait_tensorcnt)
    #define HAS_TDM 1
  #else
    #define HAS_TDM 0
  #endif
#else
  #define HAS_TDM 1   // host pass: parse-only, kernels never execute here
#endif

__device__ __forceinline__ void tdm_load_tile_bf16_pad80(
    const unsigned short* gsrc, unsigned lds_byte_off,
    unsigned tensor_w, unsigned tensor_h,
    unsigned tile_w, unsigned tile_h, unsigned stride_elems)
{
#if defined(__HIP_DEVICE_COMPILE__) && HAS_TDM
  const unsigned long long ga = (unsigned long long)(uintptr_t)gsrc;
  u32x4 g0;
  g0[0] = 1u;                                        // count=1, user D#
  g0[1] = lds_byte_off;                              // lds_addr (bytes)
  g0[2] = (unsigned)(ga & 0xffffffffu);              // global_addr[31:0]
  g0[3] = (unsigned)((ga >> 32) & 0x01ffffffu)       // global_addr[56:32]
        | 0x80000000u;                               // type = 2 ("image")
  i32x8 g1;
  g1[0] = (int)((1u << 16)                           // data_size = 2B
              | (1u << 20)                           // pad_enable
              | (3u << 22)                           // pad_interval: 16 DWORDs
              | (3u << 25));                         // pad_amount:    4 DWORDs
  g1[1] = (int)(tensor_w << 16);                     // tensor_dim0[15:0]
  g1[2] = (int)((tensor_w >> 16) | (tensor_h << 16));// dim0[31:16] | dim1[15:0]
  g1[3] = (int)((tensor_h >> 16) | (tile_w << 16));  // dim1[31:16] | tile_dim0
  g1[4] = (int)(tile_h & 0xffffu);                   // tile_dim1 (tile_dim2=0)
  g1[5] = (int)stride_elems;                         // tensor_dim0_stride lo32
  g1[6] = 0;
  g1[7] = 0;
  const i32x4 z4 = {0, 0, 0, 0};                     // 2D: groups 2/3 unused
  const i32x8 z8 = {0, 0, 0, 0, 0, 0, 0, 0};
  __builtin_amdgcn_tensor_load_to_lds(g0, g1, z4, z4, z8, 0);
#else
  (void)gsrc; (void)lds_byte_off; (void)tensor_w; (void)tensor_h;
  (void)tile_w; (void)tile_h; (void)stride_elems;
#endif
}

__device__ __forceinline__ void tdm_wait_all() {
#if defined(__HIP_DEVICE_COMPILE__) && HAS_TDM
  __builtin_amdgcn_s_wait_tensorcnt(0);
#endif
}

// ---------------------------------------------------------------------------
// GEMM: C[M,N] = A[M,K] * W[N,K]^T    (torch Linear convention)
// A is f32 or bf16 (raw ushort); W always f32; out f32 or bf16.
// Block tile 128x128x32, 256 threads = 8 waves in 2x4 arrangement.
// bf16-A tiles are DMA'd by the TDM when available.
// ---------------------------------------------------------------------------
constexpr int BM = 128, BN = 128, BK = 32;
constexpr int LSTR = 40;   // ushorts per LDS row: 80B => 16B-aligned, bank-spread

__device__ __forceinline__ void stage16(const float* __restrict__ src,
                                        unsigned short* dst) {
  const float4* s4 = (const float4*)src;
#pragma unroll
  for (int i = 0; i < 4; ++i) {
    float4 v = s4[i];
    ((uint32_t*)dst)[2*i]     = pack2_bf16(v.x, v.y);
    ((uint32_t*)dst)[2*i + 1] = pack2_bf16(v.z, v.w);
  }
}
__device__ __forceinline__ void stage16(const unsigned short* __restrict__ src,
                                        unsigned short* dst) {
  const uint4* s = (const uint4*)src;
  ((uint4*)dst)[0] = s[0];
  ((uint4*)dst)[1] = s[1];
}

template <typename AT, bool OUT_BF16>
__global__ __launch_bounds__(256)
void gemm_xwt(const AT* __restrict__ A, const float* __restrict__ W,
              void* __restrict__ Cout, int M, int N, int K)
{
  __shared__ unsigned short As[BM * LSTR];
  __shared__ unsigned short Bs[BN * LSTR];

  constexpr bool TDM_A = (HAS_TDM != 0) && (sizeof(AT) == 2);

  const int tid  = threadIdx.x;
  const int lane = tid & 31;
  const int wave = tid >> 5;
  const int wm   = wave >> 2;      // 0..1  -> 64 output rows
  const int wn   = wave & 3;       // 0..3  -> 32 output cols
  const int half = lane >> 4;      // lane group (wave32 halves)
  const int l16  = lane & 15;
  const int m0   = blockIdx.y * BM;
  const int n0   = blockIdx.x * BN;

  // staging: each thread converts/copies 16 consecutive elements (half a row)
  const int srow = tid >> 1;            // 0..127
  const int scol = (tid & 1) * 16;      // 0 / 16

  v8f acc[4][2];
#pragma unroll
  for (int i = 0; i < 4; ++i)
#pragma unroll
    for (int j = 0; j < 2; ++j)
#pragma unroll
      for (int r = 0; r < 8; ++r) acc[i][j][r] = 0.0f;

  for (int kk = 0; kk < K; kk += BK) {
    if constexpr (TDM_A) {
      if (wave == 0) {
        const unsigned lds_off = (unsigned)(uintptr_t)(void*)&As[0];
        tdm_load_tile_bf16_pad80((const unsigned short*)A + (size_t)m0 * K + kk,
                                 lds_off, (unsigned)K, (unsigned)M,
                                 BK, BM, (unsigned)K);
      }
    } else {
      stage16(A + (size_t)(m0 + srow) * K + kk + scol, &As[srow * LSTR + scol]);
    }
    stage16(W + (size_t)(n0 + srow) * K + kk + scol, &Bs[srow * LSTR + scol]);

    if (kk + BK < K) {   // prefetch next k-step tiles (global_prefetch_b8)
      __builtin_prefetch(A + (size_t)(m0 + srow) * K + kk + BK + scol, 0, 3);
      __builtin_prefetch(W + (size_t)(n0 + srow) * K + kk + BK + scol, 0, 3);
    }

    if constexpr (TDM_A) {
      if (wave == 0) tdm_wait_all();
    }
    __syncthreads();

    // A fragment (16x32 bf16): per lane row m=l16, k-chunks {half*8.. , +16}
    Frag af[4];
#pragma unroll
    for (int ms = 0; ms < 4; ++ms) {
      const unsigned short* p = &As[(wm*64 + ms*16 + l16) * LSTR + half*8];
      af[ms].q[0] = *(const uint4*)(p);
      af[ms].q[1] = *(const uint4*)(p + 16);
    }
    // B fragment (32x16 bf16 = W^T): per lane col n=l16, k-chunks {half*8.. , +16}
    Frag bfg[2];
#pragma unroll
    for (int ns = 0; ns < 2; ++ns) {
      const unsigned short* p = &Bs[(wn*32 + ns*16 + l16) * LSTR + half*8];
      bfg[ns].q[0] = *(const uint4*)(p);
      bfg[ns].q[1] = *(const uint4*)(p + 16);
    }
#pragma unroll
    for (int ms = 0; ms < 4; ++ms)
#pragma unroll
      for (int ns = 0; ns < 2; ++ns)
        acc[ms][ns] = wmma_bf16(af[ms], bfg[ns], acc[ms][ns]);
    __syncthreads();
  }

  // C layout: vgpr r, lane -> (m = r + 8*half, n = l16) within each 16x16 tile
#pragma unroll
  for (int ms = 0; ms < 4; ++ms)
#pragma unroll
    for (int ns = 0; ns < 2; ++ns) {
      const int n = n0 + wn*32 + ns*16 + l16;
#pragma unroll
      for (int r = 0; r < 8; ++r) {
        const int m = m0 + wm*64 + ms*16 + half*8 + r;
        const float v = acc[ms][ns][r];
        if constexpr (OUT_BF16)
          ((unsigned short*)Cout)[(size_t)m * N + n] = f32_to_bf16u(v);
        else
          ((float*)Cout)[(size_t)m * N + n] = v;
      }
    }
}

// ---------------------------------------------------------------------------
// Flash attention: one wave per (b, h, 16-query tile).
// S^T = K*Q^T so each lane owns ONE query column -> per-lane online softmax
// (one ds_swizzle SWAPX16 merges the wave halves). The S^T C-tile register
// layout equals the transposed A-fragment layout, so P feeds the P*V WMMA
// directly after bf16 packing. V is transposed into LDS (v_perm pair-packing,
// b32 stores) so P*V B-fragments are plain 16B ds_load_b128 reads.
// ---------------------------------------------------------------------------
__global__ __launch_bounds__(32)
void attn_flash(const unsigned short* __restrict__ qkv,
                unsigned short* __restrict__ attn)
{
  constexpr int S = 2048, E = 2048, H = 16, D = 128, LD = 3 * E;
  constexpr int VSTR = 40;                  // Vt row stride (ushorts)
  const int bid  = blockIdx.x;
  const int qt   = bid & (S / 16 - 1);      // 0..127
  const int h    = (bid >> 7) & (H - 1);
  const int b    = bid >> 11;
  const int lane = threadIdx.x & 31;
  const int half = lane >> 4;
  const int l16  = lane & 15;
  const int q0   = qt * 16;
  const int qrow = q0 + l16;                // this lane's query index

  __shared__ unsigned short Vt[D * VSTR];   // V^T chunk: [d][key], padded rows
  __shared__ float stats[16];               // per-query alpha / l broadcast

  const unsigned short* Qb = qkv + (size_t)b * S * LD + h * D;
  const unsigned short* Kb = Qb + E;
  const unsigned short* Vb = Qb + 2 * E;

  // Q^T B-fragments (loaded once): lane = column q, d-chunks {half*8.. , +16}
  Frag qf[4];
#pragma unroll
  for (int dc = 0; dc < 4; ++dc) {
    const unsigned short* p = Qb + (size_t)(q0 + l16) * LD + dc*32 + half*8;
    qf[dc].q[0] = *(const uint4*)(p);
    qf[dc].q[1] = *(const uint4*)(p + 16);
  }

  v8f o[8];                                 // O tile accum: 8 d-tiles of 16
#pragma unroll
  for (int dt = 0; dt < 8; ++dt)
#pragma unroll
    for (int r = 0; r < 8; ++r) o[dt][r] = 0.0f;

  float m_run = -1e30f, l_run = 0.0f;
  const float scale = 0.08838834764831845f; // 1/sqrt(128)
  const int nchunk = (q0 + 16 + 31) >> 5;   // causal: keys 0..q0+15

  // transpose-staging role: lane owns key pair (p2, p2+1), d-half dh
  const int p2 = (lane & 15) * 2;
  const int dh = (lane >> 4) * 64;

  for (int c = 0; c < nchunk; ++c) {
    const int k0 = c * 32;

    if (c + 1 < nchunk) {                   // prefetch next chunk K/V rows
      __builtin_prefetch(Kb + (size_t)(k0 + 32 + l16) * LD, 0, 3);
      __builtin_prefetch(Vb + (size_t)(k0 + 32 + p2) * LD + dh, 0, 3);
    }

    // stage V chunk transposed: pack {V[k0+p2][d], V[k0+p2+1][d]} per b32
    {
      const uint4* rA = (const uint4*)(Vb + (size_t)(k0 + p2)     * LD + dh);
      const uint4* rB = (const uint4*)(Vb + (size_t)(k0 + p2 + 1) * LD + dh);
#pragma unroll
      for (int i = 0; i < 8; ++i) {
        const uint4 a = rA[i], bq = rB[i];
        const uint32_t aw[4] = {a.x, a.y, a.z, a.w};
        const uint32_t bw[4] = {bq.x, bq.y, bq.z, bq.w};
        const int dr = dh + 8 * i;
#pragma unroll
        for (int wd = 0; wd < 4; ++wd) {
          const uint32_t lo = __builtin_amdgcn_perm(bw[wd], aw[wd], 0x05040100u);
          const uint32_t hi = __builtin_amdgcn_perm(bw[wd], aw[wd], 0x07060302u);
          *(uint32_t*)&Vt[(dr + 2*wd    ) * VSTR + p2] = lo;
          *(uint32_t*)&Vt[(dr + 2*wd + 1) * VSTR + p2] = hi;
        }
      }
    }

    // S^T tiles (keys k0..+15 and k0+16..+31): A = K rows, B = Q^T
    v8f st0, st1;
#pragma unroll
    for (int r = 0; r < 8; ++r) { st0[r] = 0.0f; st1[r] = 0.0f; }
#pragma unroll
    for (int dc = 0; dc < 4; ++dc) {
      Frag kf0, kf1;
      const unsigned short* pk0 = Kb + (size_t)(k0 + l16) * LD + dc*32 + half*8;
      kf0.q[0] = *(const uint4*)(pk0);
      kf0.q[1] = *(const uint4*)(pk0 + 16);
      const unsigned short* pk1 = pk0 + (size_t)16 * LD;
      kf1.q[0] = *(const uint4*)(pk1);
      kf1.q[1] = *(const uint4*)(pk1 + 16);
      st0 = wmma_bf16(kf0, qf[dc], st0);
      st1 = wmma_bf16(kf1, qf[dc], st1);
    }

    // scale + causal mask + per-lane online softmax update
    float s0[8], s1[8], mx = -1e30f;
#pragma unroll
    for (int r = 0; r < 8; ++r) {
      const int key0 = k0 + half*8 + r;     // C layout: m = r + 8*half
      const int key1 = key0 + 16;
      float av = st0[r] * scale; if (key0 > qrow) av = -1e30f;
      float dv = st1[r] * scale; if (key1 > qrow) dv = -1e30f;
      s0[r] = av; s1[r] = dv;
      mx = fmaxf(mx, fmaxf(av, dv));
    }
    mx = fmaxf(mx, swap16f(mx));            // merge lane/lane+16 halves
    const float m_new = fmaxf(m_run, mx);
    const float alpha = __expf(m_run - m_new);
    float p0v[8], p1v[8], ps = 0.0f;
#pragma unroll
    for (int r = 0; r < 8; ++r) {
      p0v[r] = __expf(s0[r] - m_new);
      p1v[r] = __expf(s1[r] - m_new);
      ps += p0v[r] + p1v[r];
    }
    ps += swap16f(ps);
    l_run = l_run * alpha + ps;
    m_run = m_new;

    // P A-fragment (16q x 32k): S^T C-layout transposes exactly onto it
    Frag pf;
#pragma unroll
    for (int j = 0; j < 4; ++j) {
      pf.w[j]     = pack2_bf16(p0v[2*j], p0v[2*j + 1]);
      pf.w[4 + j] = pack2_bf16(p1v[2*j], p1v[2*j + 1]);
    }

    // broadcast per-query alpha to O-layout rows via LDS stats
    __syncthreads();
    if (half == 0) stats[l16] = alpha;      // lanes 0..15 own queries q0+l16
    __syncthreads();                        // also orders Vt stores vs reads
    float ar[8];
#pragma unroll
    for (int r = 0; r < 8; ++r) ar[r] = stats[half*8 + r];
#pragma unroll
    for (int dt = 0; dt < 8; ++dt)
#pragma unroll
      for (int r = 0; r < 8; ++r) o[dt][r] *= ar[r];

    // O += P * V : B-fragment = Vt row d, key chunks {half*8.. , +16} -> 2x b128
#pragma unroll
    for (int dt = 0; dt < 8; ++dt) {
      Frag vf;
      const unsigned short* vp = &Vt[(dt*16 + l16) * VSTR + half*8];
      vf.q[0] = *(const uint4*)(vp);
      vf.q[1] = *(const uint4*)(vp + 16);
      o[dt] = wmma_bf16(pf, vf, o[dt]);
    }
    __syncthreads();                        // WAR: Vt reused next chunk
  }

  // final 1/l normalization + store bf16 to attn[B*S, E]
  if (half == 0) stats[l16] = l_run;
  __syncthreads();
  float li[8];
#pragma unroll
  for (int r = 0; r < 8; ++r) li[r] = 1.0f / stats[half*8 + r];

#pragma unroll
  for (int dt = 0; dt < 8; ++dt)
#pragma unroll
    for (int r = 0; r < 8; ++r) {
      const size_t row = (size_t)b * S + q0 + half*8 + r;
      attn[row * E + h*D + dt*16 + l16] = f32_to_bf16u(o[dt][r] * li[r]);
    }
}

// ---------------------------------------------------------------------------
// launch: qkv GEMM -> flash attention -> proj GEMM
// ---------------------------------------------------------------------------
extern "C" void kernel_launch(void* const* d_in, const int* in_sizes, int n_in,
                              void* d_out, int out_size, void* d_ws, size_t ws_size,
                              hipStream_t stream)
{
  (void)in_sizes; (void)n_in; (void)out_size; (void)ws_size;
  const float* x     = (const float*)d_in[0];   // [2,2048,2048]
  const float* Wqkv  = (const float*)d_in[1];   // [6144,2048]
  const float* Wproj = (const float*)d_in[2];   // [2048,2048]
  float* out = (float*)d_out;                   // [2,2048,2048]

  constexpr int Bc = 2, S = 2048, E = 2048;
  constexpr int M = Bc * S;                     // 4096 rows

  unsigned short* qkv  = (unsigned short*)d_ws;            // [4096, 6144] bf16
  unsigned short* attn = qkv + (size_t)M * 3 * E;          // [4096, 2048] bf16

  dim3 g1(3*E / BN, M / BM);   // 48 x 32
  gemm_xwt<float, true><<<g1, 256, 0, stream>>>(x, Wqkv, qkv, M, 3*E, E);

  attn_flash<<<dim3(Bc * 16 * (S / 16)), dim3(32), 0, stream>>>(qkv, attn);

  dim3 g2(E / BN, M / BM);     // 16 x 32
  gemm_xwt<unsigned short, false><<<g2, 256, 0, stream>>>(attn, Wproj, out, M, E, E);
}
